// GCNModel_52544629899986
// MI455X (gfx1250) — compile-verified, matrix operand order checked by reading, not run on hardware
//
#include <hip/hip_runtime.h>
#include <hip/hip_bf16.h>
#include <stdint.h>

typedef __attribute__((ext_vector_type(16))) __bf16 v16bf;
typedef __attribute__((ext_vector_type(8)))  float  v8f;

#define NPIX  7056           // 84*84
#define NB    64             // batch
#define COUT  256            // conv2/conv3 output channels
#define EPSB  1e-5f

union Frag { uint4 u[2]; v16bf v; };

// ---------------------------------------------------------------------------
// conv1: 3->128, 3x3 SAME, fused BN+ReLU, f32 in (NCHW) -> bf16 out [b][p][c]
// ---------------------------------------------------------------------------
__global__ void __launch_bounds__(256)
conv1_bn_relu(const float* __restrict__ x,  const float* __restrict__ w,
              const float* __restrict__ bias,
              const float* __restrict__ g,  const float* __restrict__ be,
              const float* __restrict__ mn, const float* __restrict__ vr,
              __bf16* __restrict__ out) {
  long idx = (long)blockIdx.x * blockDim.x + threadIdx.x;
  if (idx >= (long)NB * NPIX * 128) return;
  int c = (int)(idx & 127);
  long t = idx >> 7;
  int p = (int)(t % NPIX);
  int b = (int)(t / NPIX);
  int y = p / 84, xc = p - (p / 84) * 84;
  float acc = bias[c];
  #pragma unroll
  for (int i = 0; i < 3; ++i) {
    #pragma unroll
    for (int ky = 0; ky < 3; ++ky) {
      int yy = y + ky - 1;
      if (yy < 0 || yy >= 84) continue;
      #pragma unroll
      for (int kx = 0; kx < 3; ++kx) {
        int xx = xc + kx - 1;
        if (xx < 0 || xx >= 84) continue;
        float xv = x[(((size_t)b * 3 + i) * 84 + yy) * 84 + xx];
        float wv = w[((c * 3 + i) * 3 + ky) * 3 + kx];
        acc = fmaf(xv, wv, acc);
      }
    }
  }
  float sc  = g[c] * rsqrtf(vr[c] + EPSB);
  float val = fmaxf((acc - mn[c]) * sc + be[c], 0.f);
  out[idx] = (__bf16)val;
}

// ---------------------------------------------------------------------------
// Pack conv weights [O][I][3][3] f32 into WMMA-B fragment order:
// [kChunk][nTile(16)][lane(32)][16 bf16], k = tap*Cin + c, n = outCh.
// Lane l<16 holds column n, K = kg*8+j (j<8) / 16+kg*8+(j-8), kg = l>>4.
// ---------------------------------------------------------------------------
__global__ void __launch_bounds__(256)
pack_wfrag(const float* __restrict__ w, __bf16* __restrict__ out,
           int Cin, int total) {
  int idx = blockIdx.x * blockDim.x + threadIdx.x;
  if (idx >= total) return;
  int j    = idx & 15;
  int lane = (idx >> 4) & 31;
  int t2   = idx >> 9;            // kc*16 + nTile
  int gt   = t2 & 15;
  int kc   = t2 >> 4;
  int kg   = lane >> 4;
  int k    = kc * 32 + ((j >> 3) * 16 + kg * 8 + (j & 7));
  int n    = gt * 16 + (lane & 15);
  int tap  = k / Cin;
  int cc   = k - tap * Cin;
  out[idx] = (__bf16)w[((size_t)n * Cin + cc) * 9 + tap];
}

// ---------------------------------------------------------------------------
// Implicit-GEMM 3x3 SAME conv via v_wmma_f32_16x16x32_bf16.
// Block = 128 threads (4 waves). Wave: 16(M pixels) x 128(N channels) strip,
// 8 accumulators. grid = (441/4 M-tiles, 2 N-tiles, 64 batch).
// A fragments are loaded directly from global (two aligned 16B loads/lane,
// K-chunks of 32 never straddle a 3x3 tap because CIN % 32 == 0), with a
// one-chunk-ahead global_prefetch to hide HBM latency. B fragments come
// pre-swizzled from the pack kernel and stay L2-resident.
// POOL=false: fused BN+ReLU, store bf16 [b][p][n].
// POOL=true : fused BN+ReLU + global-average-pool -> atomicAdd into pooled.
// ---------------------------------------------------------------------------
template <int CIN, bool POOL>
__global__ void __launch_bounds__(128)
conv_wmma(const __bf16* __restrict__ act,    // [b][p][CIN] bf16
          const __bf16* __restrict__ wfrag,  // packed B fragments
          const float* __restrict__ bias,
          const float* __restrict__ g,  const float* __restrict__ be,
          const float* __restrict__ mn, const float* __restrict__ vr,
          __bf16* __restrict__ out,          // !POOL
          float* __restrict__ pooled) {      // POOL
  const int KC   = (9 * CIN) / 32;   // K chunks of 32
  int lane  = threadIdx.x & 31;
  int wv    = threadIdx.x >> 5;
  int b     = blockIdx.z;
  int mbase = blockIdx.x * 64 + wv * 16;
  int nbase = blockIdx.y * 128;
  int lm    = lane & 15;
  int kg    = lane >> 4;
  int m     = mbase + lm;
  bool mv   = (m < NPIX);
  int p     = mv ? m : 0;
  int y     = p / 84, xq = p - (p / 84) * 84;
  const __bf16* ab = act + (size_t)b * NPIX * CIN;

  v8f acc[8] = {};
  uint4 z4 = make_uint4(0u, 0u, 0u, 0u);

  for (int kc = 0; kc < KC; ++kc) {
    int k0  = kc * 32;
    int tap = k0 / CIN;              // CIN % 32 == 0 -> chunk stays in one tap
    int c0  = k0 - tap * CIN;
    int dy  = tap / 3 - 1;
    int dx  = tap - (tap / 3) * 3 - 1;
    int yy  = y + dy, xx = xq + dx;
    bool vld = mv && ((unsigned)yy < 84u) && ((unsigned)xx < 84u);

    Frag a;
    if (vld) {
      const __bf16* s = ab + ((size_t)(yy * 84 + xx)) * CIN + c0 + kg * 8;
      a.u[0] = *(const uint4*)(s);
      a.u[1] = *(const uint4*)(s + 16);
    } else {
      a.u[0] = z4; a.u[1] = z4;
    }

    // prefetch next chunk's A cacheline (global_prefetch_b8) while the
    // current chunk's WMMAs run; A streams from HBM (231 MB > 192 MB L2)
    int kn = kc + 1;
    if (kn < KC) {
      int k0n  = kn * 32;
      int tapn = k0n / CIN;
      int c0n  = k0n - tapn * CIN;
      int dyn  = tapn / 3 - 1;
      int dxn  = tapn - (tapn / 3) * 3 - 1;
      int yyn  = y + dyn, xxn = xq + dxn;
      if (mv && ((unsigned)yyn < 84u) && ((unsigned)xxn < 84u)) {
        const __bf16* sn = ab + ((size_t)(yyn * 84 + xxn)) * CIN + c0n + kg * 8;
        __builtin_prefetch(sn, 0, 0);
      }
    }

    const __bf16* wrow =
        wfrag + (((size_t)kc * (COUT / 16) + (nbase >> 4)) * 32 + lane) * 16;
    #pragma unroll
    for (int nt = 0; nt < 8; ++nt) {
      Frag bb;
      const uint4* bp = (const uint4*)(wrow + (size_t)nt * 32 * 16);
      bb.u[0] = bp[0];
      bb.u[1] = bp[1];
      acc[nt] = __builtin_amdgcn_wmma_f32_16x16x32_bf16(
          false, a.v, false, bb.v, (short)0, acc[nt], false, false);
    }
  }

  int mrow0 = kg * 8;   // C/D layout: lanes>=16 hold rows 8..15
  #pragma unroll
  for (int nt = 0; nt < 8; ++nt) {
    int n    = nbase + nt * 16 + lm;
    float sc = g[n] * rsqrtf(vr[n] + EPSB);
    float sh = be[n] - mn[n] * sc;
    float bi = bias[n];
    if (POOL) {
      float s = 0.f;
      #pragma unroll
      for (int r = 0; r < 8; ++r) {
        int mm = mbase + mrow0 + r;
        float val = fmaxf(fmaf(acc[nt][r] + bi, sc, sh), 0.f);
        if (mm < NPIX) s += val;
      }
      s *= (1.f / 7056.f);
      s += __shfl_xor(s, 16, 32);          // combine rows 0-7 with 8-15
      if (lane < 16) atomicAdd(pooled + b * COUT + n, s);
    } else {
      #pragma unroll
      for (int r = 0; r < 8; ++r) {
        int mm = mbase + mrow0 + r;
        if (mm < NPIX) {
          float val = fmaxf(fmaf(acc[nt][r] + bi, sc, sh), 0.f);
          out[((size_t)b * NPIX + mm) * COUT + n] = (__bf16)val;
        }
      }
    }
  }
}

// ---------------------------------------------------------------------------
// One attention-GCN layer in a single 256-thread block (needs global softmax
// over all E=256 edges). h: [64][256] f32. agg: global scratch [64][256].
// ---------------------------------------------------------------------------
__global__ void __launch_bounds__(256)
gcn_layer(const float* __restrict__ h,
          const int* __restrict__ row, const int* __restrict__ col, int E,
          const float* __restrict__ lw, const float* __restrict__ lb,
          const float* __restrict__ aw, const float* __restrict__ ab,
          float* __restrict__ agg, float* __restrict__ hout, int doRelu) {
  __shared__ float ew[256];
  __shared__ float red[256];
  int t = threadIdx.x;

  // edge attention logits
  if (t < E) {
    int r = row[t], c = col[t];
    float s = ab[0];
    for (int f = 0; f < 256; ++f)
      s += h[r * 256 + f] * aw[f] + h[c * 256 + f] * aw[256 + f];
    ew[t] = (s > 0.f) ? s : 0.01f * s;   // leaky_relu
  }
  __syncthreads();

  // softmax over all edges
  red[t] = (t < E) ? ew[t] : -1e30f;
  __syncthreads();
  for (int off = 128; off > 0; off >>= 1) {
    if (t < off) red[t] = fmaxf(red[t], red[t + off]);
    __syncthreads();
  }
  float mx = red[0];
  __syncthreads();
  float ex = (t < E) ? __expf(ew[t] - mx) : 0.f;
  red[t] = ex;
  __syncthreads();
  for (int off = 128; off > 0; off >>= 1) {
    if (t < off) red[t] += red[t + off];
    __syncthreads();
  }
  float inv = 1.f / red[0];
  __syncthreads();
  if (t < E) ew[t] = ex * inv;
  __syncthreads();

  // scatter-add: thread t owns feature column t
  for (int i = t; i < 64 * 256; i += 256) agg[i] = 0.f;
  __syncthreads();
  for (int e = 0; e < E; ++e)
    agg[row[e] * 256 + t] += h[col[e] * 256 + t] * ew[e];
  __syncthreads();

  // linear + residual (+relu); thread t owns output feature t
  for (int nnode = 0; nnode < 64; ++nnode) {
    float s = lb[t];
    for (int k = 0; k < 256; ++k)
      s += agg[nnode * 256 + k] * lw[t * 256 + k];
    s += h[nnode * 256 + t];
    if (doRelu) s = fmaxf(s, 0.f);
    hout[nnode * 256 + t] = s;
  }
}

__global__ void __launch_bounds__(256)
fc_kernel(const float* __restrict__ h, const float* __restrict__ w,
          const float* __restrict__ b, float* __restrict__ out) {
  int idx = blockIdx.x * blockDim.x + threadIdx.x;
  if (idx >= 64 * 50) return;
  int i = idx / 50, j = idx - (idx / 50) * 50;
  float s = b[j];
  for (int k = 0; k < 256; ++k) s += h[i * 256 + k] * w[j * 256 + k];
  out[idx] = s;
}

// ---------------------------------------------------------------------------
extern "C" void kernel_launch(void* const* d_in, const int* in_sizes, int n_in,
                              void* d_out, int out_size, void* d_ws, size_t ws_size,
                              hipStream_t stream) {
  const float* x    = (const float*)d_in[0];
  const float* c1w  = (const float*)d_in[1];
  const float* c1b  = (const float*)d_in[2];
  const float* b1g  = (const float*)d_in[3];
  const float* b1b  = (const float*)d_in[4];
  const float* b1m  = (const float*)d_in[5];
  const float* b1v  = (const float*)d_in[6];
  const float* c2w  = (const float*)d_in[7];
  const float* c2b  = (const float*)d_in[8];
  const float* b2g  = (const float*)d_in[9];
  const float* b2b  = (const float*)d_in[10];
  const float* b2m  = (const float*)d_in[11];
  const float* b2v  = (const float*)d_in[12];
  const float* c3w  = (const float*)d_in[13];
  const float* c3b  = (const float*)d_in[14];
  const float* b3g  = (const float*)d_in[15];
  const float* b3b  = (const float*)d_in[16];
  const float* b3m  = (const float*)d_in[17];
  const float* b3v  = (const float*)d_in[18];
  const float* g1lw = (const float*)d_in[19];
  const float* g1lb = (const float*)d_in[20];
  const float* g1aw = (const float*)d_in[21];
  const float* g1ab = (const float*)d_in[22];
  const float* g2lw = (const float*)d_in[23];
  const float* g2lb = (const float*)d_in[24];
  const float* g2aw = (const float*)d_in[25];
  const float* g2ab = (const float*)d_in[26];
  const float* fcw  = (const float*)d_in[27];
  const float* fcb  = (const float*)d_in[28];
  const int*   ei   = (const int*)d_in[29];
  int E = in_sizes[29] / 2;

  // workspace carve-out
  char* ws = (char*)d_ws;
  size_t off = 0;
  auto carve = [&](size_t bytes) -> char* {
    char* pp = ws + off;
    off += (bytes + 255) & ~(size_t)255;
    return pp;
  };
  __bf16* act1   = (__bf16*)carve((size_t)NB * NPIX * 128 * 2);
  __bf16* act2   = (__bf16*)carve((size_t)NB * NPIX * 256 * 2);
  __bf16* wf2    = (__bf16*)carve((size_t)294912 * 2);
  __bf16* wf3    = (__bf16*)carve((size_t)589824 * 2);
  float*  pooled = (float*)carve((size_t)NB * 256 * 4);
  float*  agg    = (float*)carve((size_t)NB * 256 * 4);
  float*  h1     = (float*)carve((size_t)NB * 256 * 4);
  float*  h2     = (float*)carve((size_t)NB * 256 * 4);

  // conv1 (direct, bf16 output in [b][p][c])
  {
    long total = (long)NB * NPIX * 128;
    conv1_bn_relu<<<(unsigned)((total + 255) / 256), 256, 0, stream>>>(
        x, c1w, c1b, b1g, b1b, b1m, b1v, act1);
  }

  // pack conv2/conv3 weights into WMMA B-fragment order
  pack_wfrag<<<(294912 + 255) / 256, 256, 0, stream>>>(c2w, wf2, 128, 294912);
  pack_wfrag<<<(589824 + 255) / 256, 256, 0, stream>>>(c3w, wf3, 256, 589824);

  // conv2 (WMMA implicit GEMM, store bf16 activations)
  dim3 gconv(111, 2, NB);
  conv_wmma<128, false><<<gconv, 128, 0, stream>>>(
      act1, wf2, c2b, b2g, b2b, b2m, b2v, act2, nullptr);

  // conv3 (WMMA implicit GEMM, fused BN+ReLU+global-average-pool)
  hipMemsetAsync(pooled, 0, (size_t)NB * 256 * 4, stream);
  conv_wmma<256, true><<<gconv, 128, 0, stream>>>(
      act2, wf3, c3b, b3g, b3b, b3m, b3v, nullptr, pooled);

  // GCN layers + FC (tiny)
  gcn_layer<<<1, 256, 0, stream>>>(pooled, ei, ei + E, E,
                                   g1lw, g1lb, g1aw, g1ab, agg, h1, 1);
  gcn_layer<<<1, 256, 0, stream>>>(h1, ei, ei + E, E,
                                   g2lw, g2lb, g2aw, g2ab, agg, h2, 1);
  fc_kernel<<<(64 * 50 + 255) / 256, 256, 0, stream>>>(h2, fcw, fcb,
                                                       (float*)d_out);
}
